// IdBase2d_17566416240796
// MI455X (gfx1250) — matrix operand order repaired and checked by reading, not compiled, not run on hardware
//
#include <hip/hip_runtime.h>
#include <hip/hip_bf16.h>

// ---------------------------------------------------------------------------
// Per-pixel 256->256->256 MLP with masked residual select, CDNA5 (gfx1250).
//  - bf16 WMMA (v_wmma_f32_16x16x32_bf16) with f32 accumulate
//  - weights pre-transposed/converted to bf16 once (prep kernel -> d_ws)
//  - per-WG weight staging via Tensor Data Mover (tensor_load_to_lds) with
//    hardware row padding, overlapped with X/H fragment staging
// ---------------------------------------------------------------------------

#define Bb   16
#define Cc   256
#define HID  256
#define Hh   128
#define Ww   128
#define HW   (Hh * Ww)          // 16384

// LDS layout (dynamic shared, 270336 bytes total):
//   wt  : bf16 [256][264]  weights transposed (W1^T then W2^T)   132 KB
//   xt  : bf16 [128][264]  input pixel tile                       66 KB
//   ht  : bf16 [128][264]  hidden tile                            66 KB
//   ybuf: f32  [128][260]  layer-2 output (aliases xt+ht region) 130 KB
#define WSTR 264
#define XSTR 264
#define YSTR 260
#define OFF_XT 135168
#define OFF_HT 202752
#define SMEM_BYTES 270336

typedef __bf16 bf16x16 __attribute__((ext_vector_type(16)));
typedef float  f32x8   __attribute__((ext_vector_type(8)));
typedef unsigned int u32x4 __attribute__((ext_vector_type(4)));
typedef int          i32x4 __attribute__((ext_vector_type(4)));
typedef int          i32x8 __attribute__((ext_vector_type(8)));

union Frag { uint4 q[2]; bf16x16 v; };

__device__ __forceinline__ unsigned short bfbits(float f) {
    unsigned u = __float_as_uint(f);
    return (unsigned short)((u + 0x7FFFu + ((u >> 16) & 1u)) >> 16);   // RNE
}
__device__ __forceinline__ __bf16 f2bf(float f) {
    unsigned short s = bfbits(f);
    __bf16 r;
    __builtin_memcpy(&r, &s, 2);
    return r;
}

// ---------------------------------------------------------------------------
// TDM: async-load a 256x256 bf16 tensor (row-major, contiguous) from global
// into LDS at byte offset lds_off, padding each 512B row by 16B so the LDS
// row stride is 264 bf16 elements (matches WSTR). D# per CDNA5 ISA ch.8.
// ---------------------------------------------------------------------------
__device__ __forceinline__ void tdm_load_w256(unsigned lds_off, const void* gptr) {
    unsigned long long ga = (unsigned long long)gptr;
    u32x4 g0;
    g0.x = 1u;                                               // count=1, user mode
    g0.y = lds_off;                                          // lds_addr[63:32]
    g0.z = (unsigned)(ga & 0xFFFFFFFFu);                     // global_addr[31:0]
    g0.w = (unsigned)((ga >> 32) & 0x1FFFFFFu) | (2u << 30); // addr[56:32] | type=2
    i32x8 g1;
    g1[0] = (int)((1u << 16)      // data_size = 2 bytes
                | (1u << 20)      // pad_enable
                | (6u << 22)      // pad_interval: every 128 DWORDs (=512B row)
                | (3u << 25));    // pad_amount: 4 DWORDs (=16B -> 264-elem rows)
    g1[1] = (int)(256u << 16);    // tensor_dim0 = 256
    g1[2] = (int)(256u << 16);    // tensor_dim1 = 256
    g1[3] = (int)(256u << 16);    // tile_dim0   = 256
    g1[4] = (int)256;             // tile_dim1 = 256, tile_dim2 = 0
    g1[5] = (int)256;             // tensor_dim0_stride = 256
    g1[6] = 0;
    g1[7] = 0;
    i32x4 gz = {0, 0, 0, 0};
#if __clang_major__ >= 23
    i32x8 gz8 = {0, 0, 0, 0, 0, 0, 0, 0};
    __builtin_amdgcn_tensor_load_to_lds(g0, g1, gz, gz, gz8, 0);
#else
    __builtin_amdgcn_tensor_load_to_lds(g0, g1, gz, gz, 0);
#endif
}

// A fragment (16x32, bf16): lane holds row M=lane%16.
// lanes 0-15: K = k0+{0..15}; lanes 16-31: K = k0+{16..31} (contiguous 32B)
__device__ __forceinline__ Frag load_a(const __bf16* base, int row, int k0, int lane) {
    int kb = k0 + ((lane & 16) ? 16 : 0);
    const char* p = (const char*)(base + row * XSTR + kb);
    Frag f;
    f.q[0] = *(const uint4*)p;
    f.q[1] = *(const uint4*)(p + 16);
    return f;
}

// B fragment (32x16, bf16): lane holds column N; Wt stored [N][K] row-major.
// lanes 0-15: K = k0+{0..15}; lanes 16-31: K = k0+{16..31} (contiguous 32B)
__device__ __forceinline__ Frag load_b(const __bf16* wt, int ncol, int k0, int lane) {
    int kb = k0 + ((lane & 16) ? 16 : 0);
    const char* p = (const char*)(wt + ncol * WSTR + kb);
    Frag f;
    f.q[0] = *(const uint4*)p;
    f.q[1] = *(const uint4*)(p + 16);
    return f;
}

__device__ __forceinline__ f32x8 wmma_bf16(const Frag& a, const Frag& b, f32x8 c) {
    return __builtin_amdgcn_wmma_f32_16x16x32_bf16(
        /*neg_a=*/false, a.v, /*neg_b=*/false, b.v,
        /*c_mod=*/(short)0, c, /*reuse_a=*/false, /*reuse_b=*/false);
}

// ---------------------------------------------------------------------------
// Prep: w1t[d][c] = bf16(W1[c][d]);  w2t[c][d] = bf16(W2[d][c])
// ---------------------------------------------------------------------------
__global__ void __launch_bounds__(256)
prep_weights(const float* __restrict__ W1, const float* __restrict__ W2,
             __bf16* __restrict__ w1t, __bf16* __restrict__ w2t) {
    const int r = blockIdx.x;      // output row (d for w1t, c for w2t)
    const int t = threadIdx.x;     // output col
    w1t[r * Cc + t] = f2bf(W1[t * HID + r]);
    w2t[r * HID + t] = f2bf(W2[t * Cc + r]);
}

__global__ void __launch_bounds__(256)
pixel_mlp_kernel(const float* __restrict__ in_feat,
                 const unsigned char* __restrict__ act_mask,
                 const float* __restrict__ b1, const float* __restrict__ b2,
                 const __bf16* __restrict__ w1t, const __bf16* __restrict__ w2t,
                 float* __restrict__ out_feat) {
    extern __shared__ char smem[];
    __bf16* wt   = (__bf16*)smem;              // [256][WSTR], LDS byte off 0
    __bf16* xt   = (__bf16*)(smem + OFF_XT);   // [128][XSTR]
    __bf16* ht   = (__bf16*)(smem + OFF_HT);   // [128][XSTR]
    float*  ybuf = (float*)(smem + OFF_XT);    // [128][YSTR] (aliases xt/ht)

    const int tid  = threadIdx.x;              // 0..255
    const int lane = tid & 31;
    const int wave = tid >> 5;                 // 0..7 -> M-tile
    const int m0   = wave * 16;

    const int g = blockIdx.x;                  // 0..2047 = (b,h)
    const int b = g >> 7;
    const int h = g & 127;

    const size_t img_base = (size_t)b * Cc * HW + (size_t)h * Ww;

    // ---- Phase 1a: kick off async W1^T -> wt (TDM, wave 0 only) ----
    if (wave == 0)
        tdm_load_w256(0u, w1t);

    // ---- Phase 1b: stage X tile bf16 [w][c] (overlaps the TDM) ----
    const int wpx   = tid & 127;
    const int cbase = (tid >> 7) * 2;          // 0 or 2
    {
        const float* xin = in_feat + img_base + wpx;
        for (int c = cbase; c < Cc; c += 4) {
            float f0 = xin[(size_t)c * HW];
            float f1 = xin[(size_t)(c + 1) * HW];
            unsigned u = (unsigned)bfbits(f0) | ((unsigned)bfbits(f1) << 16);
            *(unsigned*)((char*)xt + wpx * (XSTR * 2) + c * 2) = u;
        }
    }
    if (wave == 0)
        __builtin_amdgcn_s_wait_tensorcnt(0);
    __syncthreads();

    // ---- Layer 1: H = relu(X * W1 + b1) ----
    Frag xa[8];
#pragma unroll
    for (int ks = 0; ks < 8; ++ks)
        xa[ks] = load_a(xt, m0 + (lane & 15), ks * 32, lane);

    const int ncol_lane = lane & 15;
    const int prow_base = m0 + ((lane & 16) ? 8 : 0);

    for (int nt = 0; nt < 16; ++nt) {
        const int ncol = nt * 16 + ncol_lane;
        const float bias = b1[ncol];
        f32x8 acc = {0.f, 0.f, 0.f, 0.f, 0.f, 0.f, 0.f, 0.f};
#pragma unroll
        for (int ks = 0; ks < 8; ++ks) {
            Frag bf = load_b(wt, ncol, ks * 32, lane);
            acc = wmma_bf16(xa[ks], bf, acc);
        }
#pragma unroll
        for (int v = 0; v < 8; ++v) {
            float hv = fmaxf(acc[v] + bias, 0.f);
            ht[(prow_base + v) * XSTR + ncol] = f2bf(hv);
        }
    }
    __syncthreads();   // all wt reads + ht writes complete

    // ---- Phase 2a: async W2^T -> wt (TDM), overlapped with ha staging ----
    if (wave == 0)
        tdm_load_w256(0u, w2t);

    Frag ha[8];
#pragma unroll
    for (int ks = 0; ks < 8; ++ks)
        ha[ks] = load_a(ht, m0 + (lane & 15), ks * 32, lane);

    if (wave == 0)
        __builtin_amdgcn_s_wait_tensorcnt(0);
    __syncthreads();   // wt ready everywhere; all ht reads done before ybuf alias

    // ---- Layer 2: Y = H * W2 + b2 ----
    for (int nt = 0; nt < 16; ++nt) {
        const int ncol = nt * 16 + ncol_lane;
        const float bias = b2[ncol];
        f32x8 acc = {0.f, 0.f, 0.f, 0.f, 0.f, 0.f, 0.f, 0.f};
#pragma unroll
        for (int ks = 0; ks < 8; ++ks) {
            Frag bf = load_b(wt, ncol, ks * 32, lane);
            acc = wmma_bf16(ha[ks], bf, acc);
        }
#pragma unroll
        for (int v = 0; v < 8; ++v)
            ybuf[(prow_base + v) * YSTR + ncol] = acc[v] + bias;
    }
    __syncthreads();

    // ---- Final: masked select, coalesced along w ----
    {
        const unsigned m = act_mask[(size_t)b * HW + (size_t)h * Ww + wpx];
        const float* xin = in_feat + img_base + wpx;
        float*       yo  = out_feat + img_base + wpx;
        for (int c = cbase; c < Cc; c += 4) {
            float y0 = ybuf[wpx * YSTR + c];
            float y1 = ybuf[wpx * YSTR + c + 1];
            float x0 = xin[(size_t)c * HW];
            float x1 = xin[(size_t)(c + 1) * HW];
            yo[(size_t)c * HW]       = m ? y0 : x0;
            yo[(size_t)(c + 1) * HW] = m ? y1 : x1;
        }
    }
}

extern "C" void kernel_launch(void* const* d_in, const int* in_sizes, int n_in,
                              void* d_out, int out_size, void* d_ws, size_t ws_size,
                              hipStream_t stream) {
    const float*         in_feat = (const float*)d_in[0];
    const unsigned char* mask    = (const unsigned char*)d_in[1];
    const float*         W1      = (const float*)d_in[2];
    const float*         b1      = (const float*)d_in[3];
    const float*         W2      = (const float*)d_in[4];
    const float*         b2      = (const float*)d_in[5];
    float*               out     = (float*)d_out;

    __bf16* w1t = (__bf16*)d_ws;                    // [256][256] bf16, 128 KB
    __bf16* w2t = w1t + (size_t)Cc * HID;           // [256][256] bf16, 128 KB

    prep_weights<<<dim3(256), dim3(256), 0, stream>>>(W1, W2, w1t, w2t);

    hipFuncSetAttribute((const void*)pixel_mlp_kernel,
                        hipFuncAttributeMaxDynamicSharedMemorySize, SMEM_BYTES);

    dim3 grid(Bb * Hh);      // 2048 workgroups, one (b,h) row of 128 pixels each
    dim3 block(256);         // 8 waves; wave i owns 16-pixel M-tile i
    pixel_mlp_kernel<<<grid, block, SMEM_BYTES, stream>>>(
        in_feat, mask, b1, b2, w1t, w2t, out);
}
